// biLSTM_CRF_17274358465256
// MI455X (gfx1250) — matrix operand order repaired and checked by reading, not compiled, not run on hardware
//
#include <hip/hip_runtime.h>
#include <math.h>

// ---------------- CDNA5 WMMA types ----------------
typedef __attribute__((ext_vector_type(16))) _Float16 v16h;
typedef __attribute__((ext_vector_type(8)))  _Float16 v8h;
typedef __attribute__((ext_vector_type(8)))  float    v8f;
typedef __attribute__((ext_vector_type(4)))  int      v4i;

__device__ __forceinline__ v8f wmma_f16(v16h a, v16h b, v8f c) {
  // D = A(16x32 f16) * B(32x16 f16) + C(16x16 f32)
  return __builtin_amdgcn_wmma_f32_16x16x32_f16(
      /*neg_a=*/false, a, /*neg_b=*/false, b,
      /*c_mod=*/(short)0, c, /*reuse_a=*/false, /*reuse_b=*/false);
}

__device__ __forceinline__ float sigf(float x) { return 1.0f / (1.0f + expf(-x)); }

// ---- CDNA5 async global->LDS staging (ASYNCcnt), with portable fallback ----
#if defined(__gfx1250__) && __has_builtin(__builtin_amdgcn_global_load_async_to_lds_b128)
#define HAVE_ASYNC 1
#else
#define HAVE_ASYNC 0
#endif

__device__ __forceinline__ void stage16B(const _Float16* g, _Float16* l) {
#if HAVE_ASYNC
  // builtin signature: (int4 AS1* gsrc, int4 AS3* ldst, imm offset, imm cpol)
  __builtin_amdgcn_global_load_async_to_lds_b128(
      (__attribute__((address_space(1))) v4i*)g,
      (__attribute__((address_space(3))) v4i*)l, /*offset=*/0, /*cpol=*/0);
#else
  *(v8h*)l = *(const v8h*)g;
#endif
}

__device__ __forceinline__ void wait_stage() {
#if HAVE_ASYNC
#if __has_builtin(__builtin_amdgcn_s_wait_asynccnt)
  __builtin_amdgcn_s_wait_asynccnt(0);
#else
  asm volatile("s_wait_asynccnt 0" ::: "memory");
#endif
#endif
  __syncthreads();
}

// Problem sizes
#define Bb 64
#define Tt 512
#define Ff 768
#define Hh 256
#define G4 1024   /* 4*H */
#define Kk 32
#define MT (Bb*Tt) /* 32768 tokens */

// ---------------- workspace layout (bytes) ----------------
#define OFF_W16IH  ((size_t)0)                         // 2*1024*768 f16 = 3,145,728
#define OFF_W16HH  (OFF_W16IH + (size_t)2*G4*Ff*2)     // 2*1024*256 f16 = 1,048,576
#define OFF_W16LIN (OFF_W16HH + (size_t)2*G4*Hh*2)     // 32*512 f16     = 32,768
#define OFF_XG     (OFF_W16LIN + (size_t)Kk*2*Hh*2)    // 2*32768*1024 f16 = 134,217,728
#define OFF_HS     (OFF_XG + (size_t)2*MT*G4*2)        // 2*32768*256 f16  = 33,554,432
#define OFF_EM     (OFF_HS + (size_t)2*MT*Hh*2)        // 32768*32 f32     = 4,194,304
#define OFF_RES    (OFF_EM + (size_t)MT*Kk*4)          // 64 f32

// ---------------- stage 0: weight f32 -> f16 pack ----------------
__global__ void k_pack_f16(const float* __restrict__ src, _Float16* __restrict__ dst, int n) {
  int i = blockIdx.x * blockDim.x + threadIdx.x;
  if (i < n) dst[i] = (_Float16)src[i];
}

// ---------------- stage 1: input projection xg = x @ w_ih^T + b ----------------
// Block = 8 waves x 8 M-tiles sharing one 64-col N-group; B tile staged to LDS
// via async copies, double-buffered against the WMMA stream.
__global__ __launch_bounds__(256)
void k_input_proj(const float* __restrict__ x, const _Float16* __restrict__ w16ih,
                  const float* __restrict__ b_fwd, const float* __restrict__ b_bwd,
                  _Float16* __restrict__ xg)
{
  __shared__ __attribute__((aligned(32))) _Float16 bstage[2][64 * 32]; // 2 x 4KB

  const int tid  = threadIdx.x;
  const int lane = tid & 31;
  const int wid  = tid >> 5;                 // wave 0..7
  const int dir  = blockIdx.x >> 12;         // 4096 blocks per dir
  const int rem  = blockIdx.x & 4095;
  const int ngrp = rem & 15;
  const int mblk = rem >> 4;                 // 0..255
  const int m0   = (mblk * 8 + wid) * 16;    // this wave's token tile
  const int n0   = ngrp * 64;                // block-shared gate-col group
  const int half = lane >> 4;
  const int col  = lane & 15;

  const float*    xrow  = x + (size_t)(m0 + col) * Ff;   // A row for this lane
  const _Float16* wbase = w16ih + (size_t)dir * (G4 * Ff);
  const float*    bias  = dir ? b_bwd : b_fwd;

  // per-thread staging slice: 16B each (col scol, 8 f16 segment sseg)
  const int scol = tid >> 2, sseg = tid & 3;
  const _Float16* gsrc = wbase + (size_t)(n0 + scol) * Ff + sseg * 8;

  stage16B(gsrc + 0, &bstage[0][tid * 8]);   // prime buffer 0 (k0 = 0)

  v8f acc0 = {}, acc1 = {}, acc2 = {}, acc3 = {};
  int cur = 0;
  for (int kk = 0; kk < Ff / 32; ++kk) {
    const int k0 = kk * 32;
    // A fragment (global fp32 -> f16), issued before the stage-wait so the
    // loads overlap the barrier. A layout (matches contiguous B): K = i + 16*half.
    v16h a;
#pragma unroll
    for (int i = 0; i < 16; ++i)
      a[i] = (_Float16)xrow[k0 + i + 16 * half];

    wait_stage();                            // buf[cur] ready; prior reads done
    if (kk + 1 < Ff / 32)
      stage16B(gsrc + k0 + 32, &bstage[cur ^ 1][tid * 8]);

    const _Float16* bs = &bstage[cur][16 * half];
    v16h b0 = *(const v16h*)(bs + (      col) * 32);
    v16h b1 = *(const v16h*)(bs + (16 + col) * 32);
    v16h b2 = *(const v16h*)(bs + (32 + col) * 32);
    v16h b3 = *(const v16h*)(bs + (48 + col) * 32);
    acc0 = wmma_f16(a, b0, acc0);
    acc1 = wmma_f16(a, b1, acc1);
    acc2 = wmma_f16(a, b2, acc2);
    acc3 = wmma_f16(a, b3, acc3);
    cur ^= 1;
  }

  const float bn0 = bias[n0 + col], bn1 = bias[n0 + 16 + col];
  const float bn2 = bias[n0 + 32 + col], bn3 = bias[n0 + 48 + col];
#pragma unroll
  for (int r = 0; r < 8; ++r) {              // C layout: row = r + 8*half, col = lane&15
    const int token = m0 + r + 8 * half;
    const size_t base = (((size_t)dir << 15) + token) * G4;
    xg[base + n0 +  0 + col] = (_Float16)(acc0[r] + bn0);
    xg[base + n0 + 16 + col] = (_Float16)(acc1[r] + bn1);
    xg[base + n0 + 32 + col] = (_Float16)(acc2[r] + bn2);
    xg[base + n0 + 48 + col] = (_Float16)(acc3[r] + bn3);
  }
}

// ---------------- stage 2: persistent LSTM scan ----------------
// 8 blocks = 2 dirs x 4 batch-chunks (16 rows). h,c resident in LDS; each wave
// owns gate-col tiles jt = w + 8s so matched i/f/g/o stay in its registers.
// w_hh (f16, 0.5MB/dir) stays L2/WGP$-resident across all 512 steps.
__global__ __launch_bounds__(256)
void k_lstm(const _Float16* __restrict__ xg, const _Float16* __restrict__ w16hh,
            _Float16* __restrict__ hs)
{
  __shared__ __attribute__((aligned(32))) _Float16 h_lds[16 * Hh];
  __shared__ float c_lds[16 * Hh];
  const int lane  = threadIdx.x & 31;
  const int w     = threadIdx.x >> 5;        // wave 0..7
  const int dir   = blockIdx.x & 1;
  const int chunk = blockIdx.x >> 1;         // 0..3
  const int half  = lane >> 4;
  const int col   = lane & 15;

  for (int i = threadIdx.x; i < 16 * Hh; i += 256) { h_lds[i] = (_Float16)0.0f; c_lds[i] = 0.0f; }
  __syncthreads();

  const _Float16* whh = w16hh + (size_t)dir * (G4 * Hh);

  for (int step = 0; step < Tt; ++step) {
    const int t = dir ? (Tt - 1 - step) : step;

    // prefetch this step's xg lines while doing the GEMM
    {
      const int m = threadIdx.x & 15;
      const int token = (chunk * 16 + m) * Tt + t;
      __builtin_prefetch(xg + (((size_t)dir << 15) + token) * G4 + (threadIdx.x >> 4) * 64, 0, 0);
    }

    v8f acc[8] = {};
    for (int k0 = 0; k0 < Hh; k0 += 32) {
      v16h a;
#pragma unroll
      for (int i = 0; i < 16; ++i)
        a[i] = h_lds[col * Hh + k0 + i + 16 * half];
      const _Float16* wk = whh + k0 + 16 * half;
#pragma unroll
      for (int s = 0; s < 8; ++s) {
        const int ngate = 16 * (w + 8 * s) + col;
        v16h b = *(const v16h*)(wk + (size_t)ngate * Hh);
        acc[s] = wmma_f16(a, b, acc[s]);
      }
    }
    __syncthreads();   // everyone done reading h_lds

#pragma unroll
    for (int cs = 0; cs < 2; ++cs) {
      const int j = 16 * w + 128 * cs + col;     // column within H
#pragma unroll
      for (int r = 0; r < 8; ++r) {
        const int m     = r + 8 * half;
        const int token = (chunk * 16 + m) * Tt + t;
        const size_t gb = (((size_t)dir << 15) + token) * G4;
        float gi = acc[0 + cs][r] + (float)xg[gb +       j];
        float gf = acc[2 + cs][r] + (float)xg[gb + 256 + j];
        float gg = acc[4 + cs][r] + (float)xg[gb + 512 + j];
        float go = acc[6 + cs][r] + (float)xg[gb + 768 + j];
        float c  = c_lds[m * Hh + j];
        c = sigf(gf) * c + sigf(gi) * tanhf(gg);
        const float h = sigf(go) * tanhf(c);
        c_lds[m * Hh + j] = c;
        h_lds[m * Hh + j] = (_Float16)h;
        hs[(((size_t)dir << 15) + token) * Hh + j] = (_Float16)h;
      }
    }
    __syncthreads();   // new h visible before next step's GEMM
  }
}

// ---------------- stage 3: emissions = [h_f||h_b] @ w_lin^T + b_lin ----------------
// Whole w_lin (32KB f16) staged to LDS once per block via async copies.
__global__ __launch_bounds__(256)
void k_emissions(const _Float16* __restrict__ hs, const _Float16* __restrict__ w16lin,
                 const float* __restrict__ b_lin, float* __restrict__ em)
{
  __shared__ __attribute__((aligned(32))) _Float16 wl_lds[Kk * 2 * Hh]; // 32KB

  const int tid   = threadIdx.x;
  const int lane  = tid & 31;
  const int wv    = tid >> 5;
  const int mtile = blockIdx.x * 8 + wv;     // [0, 2048)
  const int m0    = mtile * 16;
  const int half  = lane >> 4, col = lane & 15;

  for (int i = tid; i < (Kk * 2 * Hh) / 8; i += 256)  // 2048 x 16B chunks
    stage16B(w16lin + i * 8, &wl_lds[i * 8]);
  wait_stage();

  v8f acc0 = {}, acc1 = {};
  for (int kk = 0; kk < 16; ++kk) {
    const int k0  = kk * 32;
    const int dir = k0 >> 8;                 // k<256 -> h_f, else h_b
    const int kl0 = k0 & 255;
    const _Float16* hrow = hs + (((size_t)dir << 15) + (m0 + col)) * Hh + kl0;
    v16h a;
#pragma unroll
    for (int i = 0; i < 16; ++i) a[i] = hrow[i + 16 * half];
    const _Float16* wk = &wl_lds[k0 + 16 * half];
    v16h b0 = *(const v16h*)(wk + (size_t)(col)      * (2 * Hh));
    v16h b1 = *(const v16h*)(wk + (size_t)(16 + col) * (2 * Hh));
    acc0 = wmma_f16(a, b0, acc0);
    acc1 = wmma_f16(a, b1, acc1);
  }
  const float bb0 = b_lin[col], bb1 = b_lin[16 + col];
#pragma unroll
  for (int r = 0; r < 8; ++r) {
    const int token = m0 + r + 8 * half;
    em[(size_t)token * Kk + col]      = acc0[r] + bb0;
    em[(size_t)token * Kk + 16 + col] = acc1[r] + bb1;
  }
}

// ---------------- stage 4: CRF log-likelihood (one wave32 per sequence) ----------------
__global__ __launch_bounds__(32)
void k_crf(const float* __restrict__ em, const int* __restrict__ y,
           const unsigned char* __restrict__ mask,
           const float* __restrict__ start_trans, const float* __restrict__ end_trans,
           const float* __restrict__ trans, float* __restrict__ res)
{
  __shared__ float tr[Kk * Kk];
  const int b = blockIdx.x;
  const int j = threadIdx.x;                 // tag index, lane == tag
  for (int i = j; i < Kk * Kk; i += 32) tr[i] = trans[i];
  __syncthreads();

  const float* emb = em + (size_t)b * Tt * Kk;
  const int*   yb  = y + b * Tt;
  const unsigned char* mb = mask + b * Tt;

  // ---- numerator (gold path score), lane-parallel over t ----
  float sc = 0.0f; int cnt = 0;
  for (int t = j; t < Tt; t += 32) cnt += mb[t] ? 1 : 0;
  for (int t = 1 + j; t < Tt; t += 32)
    if (mb[t]) sc += emb[t * Kk + yb[t]] + tr[yb[t - 1] * Kk + yb[t]];
  for (int m = 16; m > 0; m >>= 1) {
    sc  += __shfl_xor(sc, m, 32);
    cnt += __shfl_xor(cnt, m, 32);
  }

  // ---- forward algorithm: alpha[j] in lane j, all-to-all via shfl ----
  float alpha = start_trans[j] + emb[j];
  for (int t = 1; t < Tt; ++t) {
    const float emj = emb[t * Kk + j];
    float mx = -3.4e38f, s = 0.0f;
    for (int i = 0; i < Kk; ++i) {
      const float av = __shfl(alpha, i, 32);
      const float v  = av + tr[i * Kk + j];
      const float nm = fmaxf(mx, v);
      s  = s * expf(mx - nm) + expf(v - nm);
      mx = nm;
    }
    const float nxt = mx + logf(s) + emj;
    alpha = mb[t] ? nxt : alpha;
  }
  float v  = alpha + end_trans[j];
  float mx = v;
  for (int m = 16; m > 0; m >>= 1) mx = fmaxf(mx, __shfl_xor(mx, m, 32));
  float s = expf(v - mx);
  for (int m = 16; m > 0; m >>= 1) s += __shfl_xor(s, m, 32);
  const float denom = mx + logf(s);

  if (j == 0) {
    float score = sc + start_trans[yb[0]] + emb[yb[0]];
    score += end_trans[yb[cnt - 1]];
    res[b] = score - denom;
  }
}

__global__ __launch_bounds__(32)
void k_reduce(const float* __restrict__ res, float* __restrict__ out) {
  const int j = threadIdx.x;
  float s = res[j] + res[j + 32];
  for (int m = 16; m > 0; m >>= 1) s += __shfl_xor(s, m, 32);
  if (j == 0) out[0] = s;
}

// ---------------- launcher ----------------
extern "C" void kernel_launch(void* const* d_in, const int* in_sizes, int n_in,
                              void* d_out, int out_size, void* d_ws, size_t ws_size,
                              hipStream_t stream) {
  const float* x        = (const float*)d_in[0];
  const int*   y        = (const int*)d_in[1];
  const unsigned char* mask = (const unsigned char*)d_in[2];
  const float* w_ih_f   = (const float*)d_in[3];
  const float* w_hh_f   = (const float*)d_in[4];
  const float* b_f      = (const float*)d_in[5];
  const float* w_ih_b   = (const float*)d_in[6];
  const float* w_hh_b   = (const float*)d_in[7];
  const float* b_b      = (const float*)d_in[8];
  const float* w_lin    = (const float*)d_in[9];
  const float* b_lin    = (const float*)d_in[10];
  const float* start_tr = (const float*)d_in[11];
  const float* end_tr   = (const float*)d_in[12];
  const float* trans    = (const float*)d_in[13];

  char* ws = (char*)d_ws;
  _Float16* w16ih  = (_Float16*)(ws + OFF_W16IH);
  _Float16* w16hh  = (_Float16*)(ws + OFF_W16HH);
  _Float16* w16lin = (_Float16*)(ws + OFF_W16LIN);
  _Float16* xg     = (_Float16*)(ws + OFF_XG);
  _Float16* hs     = (_Float16*)(ws + OFF_HS);
  float*    em     = (float*)(ws + OFF_EM);
  float*    res    = (float*)(ws + OFF_RES);
  float*    out    = (float*)d_out;

  // stage 0: pack weights to f16
  k_pack_f16<<<(G4 * Ff + 255) / 256, 256, 0, stream>>>(w_ih_f, w16ih, G4 * Ff);
  k_pack_f16<<<(G4 * Ff + 255) / 256, 256, 0, stream>>>(w_ih_b, w16ih + (size_t)G4 * Ff, G4 * Ff);
  k_pack_f16<<<(G4 * Hh + 255) / 256, 256, 0, stream>>>(w_hh_f, w16hh, G4 * Hh);
  k_pack_f16<<<(G4 * Hh + 255) / 256, 256, 0, stream>>>(w_hh_b, w16hh + (size_t)G4 * Hh, G4 * Hh);
  k_pack_f16<<<(Kk * 2 * Hh + 255) / 256, 256, 0, stream>>>(w_lin, w16lin, Kk * 2 * Hh);

  // stage 1: input projections (both dirs), 65536 waves, B staged via async-LDS
  k_input_proj<<<8192, 256, 0, stream>>>(x, w16ih, b_f, b_b, xg);

  // stage 2: persistent recurrent scan (2 dirs x 4 batch chunks)
  k_lstm<<<8, 256, 0, stream>>>(xg, w16hh, hs);

  // stage 3: emissions
  k_emissions<<<256, 256, 0, stream>>>(hs, w16lin, b_lin, em);

  // stage 4: CRF per-sequence + final sum
  k_crf<<<Bb, 32, 0, stream>>>(em, y, mask, start_tr, end_tr, trans, res);
  k_reduce<<<1, 32, 0, stream>>>(res, out);
}
